// CrossAttention_85426899517868
// MI455X (gfx1250) — matrix operand order repaired and checked by reading, not compiled, not run on hardware
//
#include <hip/hip_runtime.h>
#include <math.h>

typedef __attribute__((ext_vector_type(16))) _Float16 v16h;
typedef __attribute__((ext_vector_type(8)))  float    v8f;
typedef __attribute__((ext_vector_type(4)))  unsigned int v4u;
typedef __attribute__((ext_vector_type(8)))  int      v8i;
typedef __attribute__((ext_vector_type(4)))  int      v4i;

#define HEADS  8
#define DHEAD  64
#define NTOK   2304
#define QDIM   320
#define INNER  512
#define SCALE  0.125f      // 64^-0.5
#define TSCALE 0.1875f     // 1.5 * SCALE

static __device__ __forceinline__ v8f wmma_f16(v16h a, v16h b, v8f c) {
  // D = A(16x32 f16) * B(32x16 f16) + C(16x16 f32)
  return __builtin_amdgcn_wmma_f32_16x16x32_f16(false, a, false, b, (short)0, c, false, false);
}

// ---------------------------------------------------------------------------
// TDM: issue a 2-D tensor_load_to_lds.  D# packed per CDNA5 ISA 08 §8.3/8.4:
//   group0: count=1 | lds_addr | global_addr[56:0] | type=2
//   group1: data_size=2B, tensor_dim0/1, tile_dim0/1, tensor_dim0_stride
//   groups 2/3: zero (2-D tensor)
// Dims/strides in data_size (2-byte) units.
// ---------------------------------------------------------------------------
static __device__ __forceinline__ void tdm_load_2d(
    unsigned lds_addr, const void* gptr,
    unsigned tile_d0, unsigned tile_d1,
    unsigned tensor_d0, unsigned tensor_d1, unsigned stride0) {
  unsigned long long ga = (unsigned long long)(uintptr_t)gptr;
  v4u g0;
  g0.x = 1u;                                            // count=1, user D#
  g0.y = lds_addr;                                      // lds_addr (bytes)
  g0.z = (unsigned)(ga & 0xffffffffu);                  // global_addr[31:0]
  g0.w = (unsigned)((ga >> 32) & 0x01ffffffu) | 0x80000000u;  // addr[56:32] | type=2
  v8i g1;
  g1[0] = (int)(1u << 16);                              // data_size=1 (2 bytes)
  g1[1] = (int)((tensor_d0 & 0xffffu) << 16);           // tensor_dim0[15:0] @ [63:48]
  g1[2] = (int)(((tensor_d0 >> 16) & 0xffffu) | ((tensor_d1 & 0xffffu) << 16));
  g1[3] = (int)(((tensor_d1 >> 16) & 0xffffu) | ((tile_d0 & 0xffffu) << 16));
  g1[4] = (int)(tile_d1 & 0xffffu);                     // tile_dim1, tile_dim2=0
  g1[5] = (int)stride0;                                 // tensor_dim0_stride[31:0]
  g1[6] = 0;
  g1[7] = 0;
  v4i z4 = {0, 0, 0, 0};
#if __has_include(<hip/amd_detail/amd_gfx1250_TDM.h>)
  v8i z8 = {0, 0, 0, 0, 0, 0, 0, 0};
  __builtin_amdgcn_tensor_load_to_lds(g0, g1, z4, z4, z8, 0);
#else
  __builtin_amdgcn_tensor_load_to_lds(g0, g1, z4, z4, 0);
#endif
}

// ---------------------------------------------------------------------------
// Kernel 1: q projection + blend.  qb[h,i,d] = 0.5*q_inj + 0.5*(x[i,:] @ Wq[:,h*64+d])
// ---------------------------------------------------------------------------
__global__ void __launch_bounds__(256) k_proj_blend(
    const float* __restrict__ x, const float* __restrict__ Wq,
    const float* __restrict__ qinj, _Float16* __restrict__ qb) {
  __shared__ float xr[QDIM];
  int i = blockIdx.x;
  int c = blockIdx.y * 256 + threadIdx.x;          // 0..511
  for (int t = threadIdx.x; t < QDIM; t += 256) xr[t] = x[i * QDIM + t];
  __syncthreads();
  float acc = 0.f;
  for (int kk = 0; kk < QDIM; ++kk) acc = fmaf(xr[kk], Wq[kk * INNER + c], acc);
  int h = c >> 6, d = c & 63;
  size_t idx = ((size_t)h * NTOK + i) * DHEAD + d;
  qb[idx] = (_Float16)(0.5f * acc + 0.5f * qinj[idx]);
}

// ---------------------------------------------------------------------------
// Kernel 2a: f32 -> f16 convert (row-major copies)
// ---------------------------------------------------------------------------
__global__ void k_f32_to_f16(const float* __restrict__ s, _Float16* __restrict__ d, int n) {
  int i = blockIdx.x * blockDim.x + threadIdx.x;
  if (i < n) d[i] = (_Float16)s[i];
}

// ---------------------------------------------------------------------------
// Kernel 2b: f32 [h][j][d] -> f16 transposed [h][d][j]  (LDS-tiled, 64x64)
// ---------------------------------------------------------------------------
__global__ void __launch_bounds__(256) k_transpose_f16(
    const float* __restrict__ src, _Float16* __restrict__ dstT) {
  __shared__ _Float16 t[64][65];                  // +1 pad: avoid bank conflicts
  int h = blockIdx.y;
  int j0 = blockIdx.x * 64;
  int tid = threadIdx.x;
  int r = tid >> 2, q = tid & 3;
  const float* S = src + ((size_t)h * NTOK + j0 + r) * DHEAD + q * 16;
#pragma unroll
  for (int i = 0; i < 16; ++i) t[r][q * 16 + i] = (_Float16)S[i];
  __syncthreads();
  _Float16* D = dstT + ((size_t)h * DHEAD + r) * NTOK + j0 + q * 16;
#pragma unroll
  for (int i = 0; i < 16; ++i) D[i] = t[q * 16 + i][r];
}

// ---------------------------------------------------------------------------
// Kernel 3: row statistics via WMMA.
//   rmin[h,i] = SCALE * min_j (qc[i] . kc[j])     (row min of cc_sim)
//   rmax[h,i] =         max_j (qc[i] . k[j])      (row max of max_sim)
// ---------------------------------------------------------------------------
__global__ void __launch_bounds__(256) k_rowstats(
    const _Float16* __restrict__ qc, const _Float16* __restrict__ kk,
    const _Float16* __restrict__ kc, float* __restrict__ rmin, float* __restrict__ rmax) {
  __shared__ float lmin[2][64];
  __shared__ float lmax[2][64];
  int tid = threadIdx.x, lane = tid & 31, w = tid >> 5;
  int rt = w & 3, ch = w >> 2;
  int h = blockIdx.y;
  int i0 = blockIdx.x * 64 + rt * 16;
  size_t hb = (size_t)h * NTOK * DHEAD;
  const _Float16* Q  = qc + hb;
  const _Float16* K1 = kk + hb;
  const _Float16* K2 = kc + hb;
  int m16 = lane & 15, hl = lane >> 4;
  int basA = hl * 8, basB = hl * 16;

  v16h a0, a1;
  size_t rb = (size_t)(i0 + m16) * DHEAD;
#pragma unroll
  for (int hh = 0; hh < 16; ++hh) {
    int kp = basB + hh;                           // A-matrix 16x32 f16 lane pattern
    a0[hh] = Q[rb + kp];
    a1[hh] = Q[rb + 32 + kp];
  }
  float mn[8], mx[8];
#pragma unroll
  for (int v = 0; v < 8; ++v) { mn[v] = INFINITY; mx[v] = -INFINITY; }

  for (int ct = ch; ct < NTOK / 16; ct += 2) {
    int j0 = ct * 16;
    size_t jb = (size_t)(j0 + m16) * DHEAD;
    v16h b0, b1;
#pragma unroll
    for (int hh = 0; hh < 16; ++hh) {             // B-matrix 32x16 f16 lane pattern
      b0[hh] = K2[jb + basB + hh];
      b1[hh] = K2[jb + 32 + basB + hh];
    }
    v8f acc = {};
    acc = wmma_f16(a0, b0, acc);
    acc = wmma_f16(a1, b1, acc);
#pragma unroll
    for (int v = 0; v < 8; ++v) mn[v] = fminf(mn[v], acc[v]);

#pragma unroll
    for (int hh = 0; hh < 16; ++hh) {
      b0[hh] = K1[jb + basB + hh];
      b1[hh] = K1[jb + 32 + basB + hh];
    }
    v8f acc2 = {};
    acc2 = wmma_f16(a0, b0, acc2);
    acc2 = wmma_f16(a1, b1, acc2);
#pragma unroll
    for (int v = 0; v < 8; ++v) mx[v] = fmaxf(mx[v], acc2[v]);
  }
#pragma unroll
  for (int v = 0; v < 8; ++v) {
    for (int m = 1; m < 16; m <<= 1) {
      mn[v] = fminf(mn[v], __shfl_xor(mn[v], m, 32));
      mx[v] = fmaxf(mx[v], __shfl_xor(mx[v], m, 32));
    }
  }
  if ((lane & 15) == 0) {
#pragma unroll
    for (int v = 0; v < 8; ++v) {
      lmin[ch][rt * 16 + v + 8 * hl] = mn[v];
      lmax[ch][rt * 16 + v + 8 * hl] = mx[v];
    }
  }
  __syncthreads();
  if (tid < 64) {
    int row = blockIdx.x * 64 + tid;
    rmin[h * NTOK + row] = fminf(lmin[0][tid], lmin[1][tid]) * SCALE;
    rmax[h * NTOK + row] = fmaxf(lmax[0][tid], lmax[1][tid]);
  }
}

// ---------------------------------------------------------------------------
// Kernel 4: flash attention over concatenated [style | content] keys.
// grid = (NTOK/128, HEADS); each of 8 waves owns 16 query rows, sweeps all
// 2*N keys with online softmax.  K/V tiles staged into LDS by the Tensor
// Data Mover (wave 0 issues descriptors, s_wait_tensorcnt, block barrier).
// ---------------------------------------------------------------------------
__global__ void __launch_bounds__(256) k_flash(
    const _Float16* __restrict__ qb, const _Float16* __restrict__ qc,
    const _Float16* __restrict__ kk, const _Float16* __restrict__ kc,
    const _Float16* __restrict__ vT, const _Float16* __restrict__ vcT,
    const float* __restrict__ rmin, const float* __restrict__ rmax,
    const float* __restrict__ mask, float* __restrict__ attn) {
  __shared__ _Float16 Kt[32 * DHEAD];     // 4 KB: K tile, [j][d] (32x64)
  __shared__ _Float16 Vt[DHEAD * 32];     // 4 KB: V^T tile, [d][j] (64x32)
  __shared__ _Float16 Pt[8][16 * 32];     // 1 KB/wave: P C->A layout bounce

  int tid = threadIdx.x, lane = tid & 31, w = tid >> 5;
  int h = blockIdx.y;
  int i0 = blockIdx.x * 128 + w * 16;
  size_t hb = (size_t)h * NTOK * DHEAD;
  int m16 = lane & 15, hl = lane >> 4;
  int basA = hl * 8, basB = hl * 16;
  (void)tid;

  // Q fragments (blended and content), K = 0..31 and 32..63
  v16h aq0, aq1, ac0, ac1;
  size_t rb = hb + (size_t)(i0 + m16) * DHEAD;
#pragma unroll
  for (int hh = 0; hh < 16; ++hh) {
    int kp = basB + hh;
    aq0[hh] = qb[rb + kp];  aq1[hh] = qb[rb + 32 + kp];
    ac0[hh] = qc[rb + kp];  ac1[hh] = qc[rb + 32 + kp];
  }
  // per-row gvm = (min_cc - max_ms) * m(row)
  float g[8];
#pragma unroll
  for (int v = 0; v < 8; ++v) {
    int row = i0 + v + 8 * hl;
    float mk = mask[row];
    float m = (mk < 0.5f) ? 1.0f : ((mk > 0.5f) ? -1.0f : -mk);  // threshold, *-1
    g[v] = (rmin[h * NTOK + row] - rmax[h * NTOK + row]) * m;
  }
  float mrow[8], lrow[8];
#pragma unroll
  for (int v = 0; v < 8; ++v) { mrow[v] = -INFINITY; lrow[v] = 0.f; }
  v8f o0 = {}, o1 = {}, o2 = {}, o3 = {};

  unsigned kt_lds = (unsigned)(uintptr_t)(void*)Kt;
  unsigned vt_lds = (unsigned)(uintptr_t)(void*)Vt;

  for (int phase = 0; phase < 2; ++phase) {
    const _Float16* Kg  = (phase ? kc  : kk) + hb;
    const _Float16* VTg = (phase ? vcT : vT) + hb;   // [d][j], row stride NTOK
    v16h A0 = phase ? ac0 : aq0;
    v16h A1 = phase ? ac1 : aq1;
    float lsc = phase ? SCALE : TSCALE;

    for (int jt = 0; jt < NTOK; jt += 32) {
      __syncthreads();                               // LDS consumers of prev tile done
      if (w == 0) {
        // K tile: 32 rows x 64 halfs, contiguous rows (stride 64)
        tdm_load_2d(kt_lds, Kg + (size_t)jt * DHEAD, 64u, 32u, 64u, (unsigned)NTOK, 64u);
        // V^T tile: 64 rows x 32 halfs, row stride NTOK
        tdm_load_2d(vt_lds, VTg + jt, 32u, 64u, (unsigned)NTOK, (unsigned)DHEAD, (unsigned)NTOK);
        __builtin_amdgcn_s_wait_tensorcnt(0);
      }
      __syncthreads();

      // S tiles: 16 rows x 32 cols, K=64 (2 chained WMMAs each)
      v16h b;
      v8f s0 = {}, s1 = {};
#pragma unroll
      for (int hh = 0; hh < 16; ++hh) b[hh] = Kt[m16 * DHEAD + basB + hh];
      s0 = wmma_f16(A0, b, s0);
#pragma unroll
      for (int hh = 0; hh < 16; ++hh) b[hh] = Kt[m16 * DHEAD + 32 + basB + hh];
      s0 = wmma_f16(A1, b, s0);
#pragma unroll
      for (int hh = 0; hh < 16; ++hh) b[hh] = Kt[(16 + m16) * DHEAD + basB + hh];
      s1 = wmma_f16(A0, b, s1);
#pragma unroll
      for (int hh = 0; hh < 16; ++hh) b[hh] = Kt[(16 + m16) * DHEAD + 32 + basB + hh];
      s1 = wmma_f16(A1, b, s1);

      // online softmax update (per row v + 8*hl)
      float p0[8], p1[8], alpha[8];
#pragma unroll
      for (int v = 0; v < 8; ++v) {
        float gg = phase ? 0.0f : g[v];
        float e0 = lsc * s0[v] + gg;
        float e1 = lsc * s1[v] + gg;
        float t = fmaxf(e0, e1);
        for (int m = 1; m < 16; m <<= 1) t = fmaxf(t, __shfl_xor(t, m, 32));
        float mnew = fmaxf(mrow[v], t);
        alpha[v] = __expf(mrow[v] - mnew);
        p0[v] = __expf(e0 - mnew);
        p1[v] = __expf(e1 - mnew);
        float rs = p0[v] + p1[v];
        for (int m = 1; m < 16; m <<= 1) rs += __shfl_xor(rs, m, 32);
        lrow[v] = lrow[v] * alpha[v] + rs;
        mrow[v] = mnew;
      }
#pragma unroll
      for (int v = 0; v < 8; ++v) {
        o0[v] *= alpha[v]; o1[v] *= alpha[v]; o2[v] *= alpha[v]; o3[v] *= alpha[v];
      }

      // P: C layout -> A layout via per-wave LDS bounce
      _Float16* P = Pt[w];
#pragma unroll
      for (int v = 0; v < 8; ++v) {
        P[(v + 8 * hl) * 32 + m16]      = (_Float16)p0[v];
        P[(v + 8 * hl) * 32 + 16 + m16] = (_Float16)p1[v];
      }
      __syncthreads();
      v16h pa;
#pragma unroll
      for (int hh = 0; hh < 16; ++hh)
        pa[hh] = P[m16 * 32 + basB + hh];

      // O += P (16x32) x V (32x16 per d-tile); V^T tile reads are contiguous
      v16h bv;
#pragma unroll
      for (int hh = 0; hh < 16; ++hh) bv[hh] = Vt[(m16)*32 + basB + hh];
      o0 = wmma_f16(pa, bv, o0);
#pragma unroll
      for (int hh = 0; hh < 16; ++hh) bv[hh] = Vt[(16 + m16) * 32 + basB + hh];
      o1 = wmma_f16(pa, bv, o1);
#pragma unroll
      for (int hh = 0; hh < 16; ++hh) bv[hh] = Vt[(32 + m16) * 32 + basB + hh];
      o2 = wmma_f16(pa, bv, o2);
#pragma unroll
      for (int hh = 0; hh < 16; ++hh) bv[hh] = Vt[(48 + m16) * 32 + basB + hh];
      o3 = wmma_f16(pa, bv, o3);
    }
  }

  // normalize and write out as (N, INNER) for the final projection
#pragma unroll
  for (int v = 0; v < 8; ++v) {
    float inv = 1.0f / lrow[v];
    int row = i0 + v + 8 * hl;
    float* orow = attn + (size_t)row * INNER + h * DHEAD;
    orow[m16]      = o0[v] * inv;
    orow[16 + m16] = o1[v] * inv;
    orow[32 + m16] = o2[v] * inv;
    orow[48 + m16] = o3[v] * inv;
  }
}

// ---------------------------------------------------------------------------
// Kernel 5: out = attn(N,512) @ Wo(512,320) + bo
// ---------------------------------------------------------------------------
__global__ void __launch_bounds__(320) k_outproj(
    const float* __restrict__ attn, const float* __restrict__ Wo,
    const float* __restrict__ bo, float* __restrict__ out) {
  __shared__ float ar[INNER];
  int i = blockIdx.x, c = threadIdx.x;
  for (int t = c; t < INNER; t += 320) ar[t] = attn[(size_t)i * INNER + t];
  __syncthreads();
  float acc = bo[c];
  for (int e = 0; e < INNER; ++e) acc = fmaf(ar[e], Wo[e * QDIM + c], acc);
  out[(size_t)i * QDIM + c] = acc;
}

// ---------------------------------------------------------------------------
extern "C" void kernel_launch(void* const* d_in, const int* in_sizes, int n_in,
                              void* d_out, int out_size, void* d_ws, size_t ws_size,
                              hipStream_t stream) {
  const float* x    = (const float*)d_in[0];
  const float* Wq   = (const float*)d_in[1];
  const float* Wo   = (const float*)d_in[2];
  const float* bo   = (const float*)d_in[3];
  const float* qinj = (const float*)d_in[4];
  const float* kinj = (const float*)d_in[5];
  const float* vinj = (const float*)d_in[6];
  const float* kcin = (const float*)d_in[7];
  const float* vcin = (const float*)d_in[8];
  const float* mask = (const float*)d_in[9];
  float* out = (float*)d_out;

  const size_t HH = (size_t)HEADS * NTOK * DHEAD;   // 1,179,648 elems
  _Float16* qb  = (_Float16*)d_ws;
  _Float16* qc  = qb + HH;
  _Float16* kk  = qc + HH;
  _Float16* kc  = kk + HH;
  _Float16* vTp = kc + HH;                          // [h][d][j] transposed
  _Float16* vcTp= vTp + HH;
  float* rmin = (float*)(vcTp + HH);
  float* rmax = rmin + HEADS * NTOK;
  float* attn = rmax + HEADS * NTOK;                // NTOK*INNER floats

  // q projection + blend
  k_proj_blend<<<dim3(NTOK, 2), 256, 0, stream>>>(x, Wq, qinj, qb);

  // f32 -> f16 converts (row-major) and transposed V / V_cnt
  int n = (int)HH;
  int blk = (n + 255) / 256;
  k_f32_to_f16<<<blk, 256, 0, stream>>>(qinj, qc, n);
  k_f32_to_f16<<<blk, 256, 0, stream>>>(kinj, kk, n);
  k_f32_to_f16<<<blk, 256, 0, stream>>>(kcin, kc, n);
  k_transpose_f16<<<dim3(NTOK / 64, HEADS), 256, 0, stream>>>(vinj, vTp);
  k_transpose_f16<<<dim3(NTOK / 64, HEADS), 256, 0, stream>>>(vcin, vcTp);

  // pass 1: row min/max stats via WMMA
  k_rowstats<<<dim3(NTOK / 64, HEADS), 256, 0, stream>>>(qc, kk, kc, rmin, rmax);

  // pass 2: flash attention over concatenated keys (WMMA + TDM staging)
  k_flash<<<dim3(NTOK / 128, HEADS), 256, 0, stream>>>(qb, qc, kk, kc, vTp, vcTp,
                                                       rmin, rmax, mask, attn);

  // output projection
  k_outproj<<<NTOK, 320, 0, stream>>>(attn, Wo, bo, out);
}
